// BiaffineSpanHead_14998025797667
// MI455X (gfx1250) — matrix operand order repaired and checked by reading, not compiled, not run on hardware
//
#include <hip/hip_runtime.h>
#include <hip/hip_bf16.h>

// ---- problem constants (reference setup_inputs: B=8, L=2048, d=1024, K=30) ----
#define BB 8
#define LL 2048
#define DD 1024
#define KK 30
#define NEGV (-1000000000.0f)

typedef __attribute__((ext_vector_type(8)))  float  v8f;
typedef __attribute__((ext_vector_type(4)))  float  v4f;
typedef __attribute__((ext_vector_type(16))) __bf16 v16bf;
typedef __attribute__((ext_vector_type(8)))  __bf16 v8bf;
typedef __attribute__((ext_vector_type(4)))  __bf16 v4bf;

__device__ __forceinline__ v16bf cat16(v8bf lo, v8bf hi) {
  return __builtin_shufflevector(lo, hi, 0,1,2,3,4,5,6,7,8,9,10,11,12,13,14,15);
}
__device__ __forceinline__ v8f wmma_bf16(v16bf A, v16bf Bm, v8f C) {
  // D = A(16x32 bf16) * B(32x16 bf16) + C(16x16 f32)
  return __builtin_amdgcn_wmma_f32_16x16x32_bf16(false, A, false, Bm, (short)0, C, false, false);
}

// ---------------- prep kernel 1: H (f32) -> H (bf16) ----------------
__global__ __launch_bounds__(256) void conv_h_kernel(const float* __restrict__ H,
                                                     unsigned short* __restrict__ Hbf_us,
                                                     int n4) {
  __bf16* Hbf = reinterpret_cast<__bf16*>(Hbf_us);
  int i = blockIdx.x * 256 + threadIdx.x;
  if (i >= n4) return;
  v4f f = *(const v4f*)(H + (size_t)i * 4);
  v4bf o;
  o.x = (__bf16)f.x; o.y = (__bf16)f.y; o.z = (__bf16)f.z; o.w = (__bf16)f.w;
  *(v4bf*)(Hbf + (size_t)i * 4) = o;
}

// ---------------- prep kernel 2: U (f32, k-major) -> U^T (bf16, n-major) ----------------
__global__ __launch_bounds__(256) void transpose_u_kernel(const float* __restrict__ U,
                                                          unsigned short* __restrict__ Ut_us) {
  __bf16* Ut = reinterpret_cast<__bf16*>(Ut_us);
  __shared__ float tile[32][33];
  const int kx = blockIdx.x * 32;           // k base
  const int ny = blockIdx.y * 32;           // n base
  const int tx = threadIdx.x & 31;
  const int ty = threadIdx.x >> 5;          // 8 rows of 32
  for (int r = ty; r < 32; r += 8)
    tile[r][tx] = U[(size_t)(kx + r) * DD + (ny + tx)];
  __syncthreads();
  for (int r = ty; r < 32; r += 8)
    Ut[(size_t)(ny + r) * DD + (kx + tx)] = (__bf16)tile[tx][r];
}

// ---------------- prep kernel 3: a = H@w1, c = H@w2 (f32) ----------------
__global__ __launch_bounds__(256) void rowdot_kernel(const float* __restrict__ H,
                                                     const float* __restrict__ w,
                                                     float* __restrict__ arow,
                                                     float* __restrict__ crow) {
  __shared__ float r1[256];
  __shared__ float r2[256];
  const size_t row = blockIdx.x;
  float s1 = 0.f, s2 = 0.f;
  for (int i = threadIdx.x; i < DD; i += 256) {
    float h = H[row * DD + i];
    s1 += h * w[i];
    s2 += h * w[DD + i];
  }
  r1[threadIdx.x] = s1; r2[threadIdx.x] = s2;
  __syncthreads();
  for (int s = 128; s > 0; s >>= 1) {
    if ((int)threadIdx.x < s) {
      r1[threadIdx.x] += r1[threadIdx.x + s];
      r2[threadIdx.x] += r2[threadIdx.x + s];
    }
    __syncthreads();
  }
  if (threadIdx.x == 0) { arow[row] = r1[0]; crow[row] = r2[0]; }
}

// ---------------- main fused kernel: one 16-row block per workgroup ----------------
// Phase 1: Hu_tile(16x1024) = H_tile @ U  (bf16 WMMA, f32 acc), Hu kept in LDS (bf16).
// Phase 2: P(16x48) = Hu_tile @ Hpad[l0..l0+47]^T  (banded product), ds_add_f32 reduce.
// Epilogue: out[b,l,off] = band(P) + a + c_shift + bias, masked.
__global__ __launch_bounds__(256) void biaffine_main_kernel(const unsigned short* __restrict__ Hbf_us,
                                                            const unsigned short* __restrict__ Ut_us,
                                                            const float* __restrict__ arow,
                                                            const float* __restrict__ crow,
                                                            const int*   __restrict__ mask,
                                                            const float* __restrict__ bptr,
                                                            float* __restrict__ out) {
  const __bf16* Hbf = reinterpret_cast<const __bf16*>(Hbf_us);
  const __bf16* Ut  = reinterpret_cast<const __bf16*>(Ut_us);

  // 16 x 1024 bf16 tile, +8 element row pad (keeps 16B alignment, breaks bank conflicts)
  __shared__ __align__(16) __bf16 sH[16][DD + 8];   // phase 1: H tile; phase 2: Hu tile (reused)
  __shared__ float sP[16][52];

  const int lane = threadIdx.x & 31;   // wave32
  const int wave = threadIdx.x >> 5;   // 8 waves
  const int lrow = lane & 15;
  const int hi   = lane >> 4;          // which K-half this lane holds

  const int blocksPerBatch = LL / 16;
  const int b  = blockIdx.x / blocksPerBatch;
  const int l0 = (blockIdx.x % blocksPerBatch) * 16;
  const size_t gr0 = (size_t)b * LL + l0;

  // zero P accumulator
  for (int i = threadIdx.x; i < 16 * 52; i += 256) ((float*)sP)[i] = 0.f;

  // cooperative load of the 16 x DD H tile (bf16) into LDS
  for (int i = threadIdx.x; i < 16 * (DD / 8); i += 256) {
    int r  = i / (DD / 8);
    int cc = (i % (DD / 8)) * 8;
    *(v8bf*)&sH[r][cc] = *(const v8bf*)(Hbf + (gr0 + r) * DD + cc);
  }
  __syncthreads();

  // -------- phase 1: Hu_tile = H_tile @ U --------
  v8f acc[8] = {};
  const int n0w = wave * 128;          // each wave owns 128 columns of Hu
  for (int k = 0; k < DD; k += 32) {
    // A fragment (16x32 bf16) from LDS per ISA layout:
    // lanes 0-15: K = k+0..7 and k+16..23 ; lanes 16-31: K = k+8..15 and k+24..31
    const __bf16* ap = &sH[lrow][k + hi * 8];
    v16bf Af = cat16(*(const v8bf*)ap, *(const v8bf*)(ap + 16));
#pragma unroll
    for (int c = 0; c < 8; ++c) {
      // B fragment (32x16 bf16): lane n holds column n0+c*16+n, K = k+8*hi+0..7 and k+16+8*hi+0..7 of U^T
      const __bf16* bp = Ut + (size_t)(n0w + c * 16 + lrow) * DD + k + hi * 8;
      v16bf Bf = cat16(*(const v8bf*)bp, *(const v8bf*)(bp + 16));
      acc[c] = wmma_bf16(Af, Bf, acc[c]);
    }
  }
  __syncthreads();   // all reads of sH (H tile) complete

  // store Hu tile (bf16) into the same LDS buffer.
  // C/D layout: VGPR e -> row M = e + 8*hi, lane -> column N.
  {
    const int mb = hi * 8;
#pragma unroll
    for (int c = 0; c < 8; ++c) {
      int nn = n0w + c * 16 + lrow;
#pragma unroll
      for (int e = 0; e < 8; ++e) sH[mb + e][nn] = (__bf16)acc[c][e];
    }
  }
  __syncthreads();

  // -------- phase 2: P = Hu_tile @ Hpad[l0 .. l0+47]^T, K split across waves --------
  v8f pacc[3] = {};
  for (int kk = 0; kk < 4; ++kk) {
    const int k = wave * 128 + kk * 32;
    const __bf16* ap = &sH[lrow][k + hi * 8];        // A from Hu tile
    v16bf Af = cat16(*(const v8bf*)ap, *(const v8bf*)(ap + 16));
#pragma unroll
    for (int nt = 0; nt < 3; ++nt) {
      const int r = l0 + nt * 16 + lrow;             // H row index within batch (column of B)
      v16bf Bf = {};
      if (r < LL) {                                  // zero-padding past sequence end
        const __bf16* bp = Hbf + ((size_t)b * LL + r) * DD + k + hi * 8;
        Bf = cat16(*(const v8bf*)bp, *(const v8bf*)(bp + 16));
      }
      pacc[nt] = wmma_bf16(Af, Bf, pacc[nt]);
    }
  }
  {
    const int mb = hi * 8;
#pragma unroll
    for (int nt = 0; nt < 3; ++nt) {
      int col = nt * 16 + lrow;
#pragma unroll
      for (int e = 0; e < 8; ++e) atomicAdd(&sP[mb + e][col], pacc[nt][e]);  // ds_add_f32
    }
  }
  __syncthreads();

  // -------- epilogue: band extract + a + c_shift + bias, mask to NEG --------
  const float bias = bptr[0];
  for (int idx = threadIdx.x; idx < 16 * KK; idx += 256) {
    const int row = idx / KK;
    const int off = idx % KK;
    const int l   = l0 + row;
    const size_t gr = gr0 + row;
    float val = sP[row][row + off] + arow[gr] + bias;
    int m = mask[gr];
    const int l2 = l + off;
    if (l2 < LL) {
      val += crow[(size_t)b * LL + l2];
      m *= mask[(size_t)b * LL + l2];
    } else {
      m = 0;
    }
    out[gr * KK + off] = (m == 0) ? NEGV : val;
  }
}

// ---------------- launch ----------------
extern "C" void kernel_launch(void* const* d_in, const int* in_sizes, int n_in,
                              void* d_out, int out_size, void* d_ws, size_t ws_size,
                              hipStream_t stream) {
  const float* H    = (const float*)d_in[0];
  const int*   mask = (const int*)  d_in[1];
  const float* U    = (const float*)d_in[2];
  const float* w    = (const float*)d_in[3];
  const float* bias = (const float*)d_in[4];
  float* out = (float*)d_out;

  // workspace layout: Hbf (32 MB) | U^T bf16 (2 MB) | a (64 KB) | c (64 KB)
  char* ws = (char*)d_ws;
  unsigned short* Hbf = (unsigned short*)ws;
  unsigned short* Ut  = (unsigned short*)(ws + (size_t)BB * LL * DD * 2);
  float* arow = (float*)(ws + (size_t)BB * LL * DD * 2 + (size_t)DD * DD * 2);
  float* crow = arow + (size_t)BB * LL;

  const int n4 = BB * LL * DD / 4;
  conv_h_kernel<<<(n4 + 255) / 256, 256, 0, stream>>>(H, Hbf, n4);
  transpose_u_kernel<<<dim3(DD / 32, DD / 32), 256, 0, stream>>>(U, Ut);
  rowdot_kernel<<<BB * LL, 256, 0, stream>>>(H, w, arow, crow);
  biaffine_main_kernel<<<BB * LL / 16, 256, 0, stream>>>(Hbf, Ut, arow, crow, mask, bias, out);
}
